// GAT_7722351198608
// MI455X (gfx1250) — matrix operand order repaired and checked by reading, not compiled, not run on hardware
//
#include <hip/hip_runtime.h>
#include <hip/hip_bf16.h>

typedef __attribute__((ext_vector_type(16))) _Float16 v16h;
typedef __attribute__((ext_vector_type(8)))  _Float16 v8h;
typedef __attribute__((ext_vector_type(8)))  float    v8f;

#define N_NODES 4096
#define F_IN    512
#define H_DIM   64
#define NH      8
#define C_CLS   40
#define C_PAD   64

__device__ __forceinline__ v8f wmma_f16(v16h a, v16h b, v8f c) {
  return __builtin_amdgcn_wmma_f32_16x16x32_f16(
      /*neg_a=*/false, a, /*neg_b=*/false, b,
      /*c_mod=*/(short)0, c, /*reuse_a=*/false, /*reuse_b=*/false);
}

// 16x32 f16 A-fragment: lanes 0-15 hold row M=lane with K=0..15;
// lanes 16-31 hold row M=lane-16 with K=16..31.
__device__ __forceinline__ v16h load_a16(const _Float16* p_lo, const _Float16* p_hi) {
  v8h lo = *(const v8h*)p_lo;
  v8h hi = *(const v8h*)p_hi;
  v16h a;
#pragma unroll
  for (int i = 0; i < 8; ++i) { a[i] = lo[i]; a[8 + i] = hi[i]; }
  return a;
}

// ---------------------------------------------------------------------------
// K0: pack/convert inputs to f16 with WMMA-friendly layouts.
// ---------------------------------------------------------------------------
__global__ void __launch_bounds__(256) k_pack(
    const float* __restrict__ seq, const float* __restrict__ W1,
    const float* __restrict__ W2, const float* __restrict__ Wo1,
    _Float16* __restrict__ seq16, _Float16* __restrict__ W1T,
    _Float16* __restrict__ W2T,   _Float16* __restrict__ Wo1T) {
  int idx = blockIdx.x * 256 + threadIdx.x;
  if (idx < N_NODES * F_IN) seq16[idx] = (_Float16)seq[idx];
  if (idx < (NH * H_DIM) * F_IN) {
    int r = idx >> 9, f = idx & 511;         // r = head*64 + h
    int head = r >> 6, h = r & 63;
    W1T[idx] = (_Float16)W1[(head * F_IN + f) * H_DIM + h];
  }
  if (idx < NH * H_DIM * H_DIM) {
    int head = idx >> 12, rem = idx & 4095;
    int ko = rem >> 6, ki = rem & 63;
    W2T[idx] = (_Float16)W2[(head * H_DIM + ki) * H_DIM + ko];
  }
  if (idx < C_PAD * F_IN) {
    int c = idx >> 9, f = idx & 511;
    Wo1T[idx] = (c < C_CLS) ? (_Float16)Wo1[f * C_CLS + c] : (_Float16)0.f;
  }
}

// ---------------------------------------------------------------------------
// K1: fts = seq16 @ W1T^T (4096x512x512), software-pipelined k-loop.
// block = 8 waves; wave w owns output columns [w*64, w*64+64).
// ---------------------------------------------------------------------------
__global__ void __launch_bounds__(256) k_gemm_fts(
    const _Float16* __restrict__ seq16, const _Float16* __restrict__ W1T,
    float* __restrict__ fts, _Float16* __restrict__ ftsT) {
  const int lane = threadIdx.x & 31;
  const int wave = threadIdx.x >> 5;
  const int i0 = blockIdx.x * 16;
  const int mlo = lane & 15, hi = lane >> 4;
  v8f acc0 = {}, acc1 = {}, acc2 = {}, acc3 = {};
  const _Float16* arow = seq16 + (i0 + mlo) * F_IN;
  const int nb = wave * 64 + mlo;
  const _Float16* brow = W1T + nb * F_IN + 16 * hi;

  // prologue: fragments for k-chunk 0
  v16h a  = load_a16(arow + 16 * hi, arow + 8 + 16 * hi);
  v16h b0 = *(const v16h*)(brow);
  v16h b1 = *(const v16h*)(brow + 16 * F_IN);
  v16h b2 = *(const v16h*)(brow + 32 * F_IN);
  v16h b3 = *(const v16h*)(brow + 48 * F_IN);
#pragma unroll 3
  for (int kt = 0; kt < F_IN / 32 - 1; ++kt) {
    const int k1 = (kt + 1) * 32;
    // load next chunk into fresh regs while current WMMAs run
    v16h an  = load_a16(arow + k1 + 16 * hi, arow + k1 + 8 + 16 * hi);
    v16h bn0 = *(const v16h*)(brow + k1);
    v16h bn1 = *(const v16h*)(brow + k1 + 16 * F_IN);
    v16h bn2 = *(const v16h*)(brow + k1 + 32 * F_IN);
    v16h bn3 = *(const v16h*)(brow + k1 + 48 * F_IN);
    acc0 = wmma_f16(a, b0, acc0);
    acc1 = wmma_f16(a, b1, acc1);
    acc2 = wmma_f16(a, b2, acc2);
    acc3 = wmma_f16(a, b3, acc3);
    a = an; b0 = bn0; b1 = bn1; b2 = bn2; b3 = bn3;
  }
  acc0 = wmma_f16(a, b0, acc0);
  acc1 = wmma_f16(a, b1, acc1);
  acc2 = wmma_f16(a, b2, acc2);
  acc3 = wmma_f16(a, b3, acc3);

#pragma unroll
  for (int v = 0; v < 8; ++v) {
    const int m = 8 * hi + v, row = i0 + m;
    float c0 = acc0[v], c1 = acc1[v], c2 = acc2[v], c3 = acc3[v];
    fts[row * 512 + nb +  0] = c0;
    fts[row * 512 + nb + 16] = c1;
    fts[row * 512 + nb + 32] = c2;
    fts[row * 512 + nb + 48] = c3;
    ftsT[(nb +  0) * N_NODES + row] = (_Float16)c0;
    ftsT[(nb + 16) * N_NODES + row] = (_Float16)c1;
    ftsT[(nb + 32) * N_NODES + row] = (_Float16)c2;
    ftsT[(nb + 48) * N_NODES + row] = (_Float16)c3;
  }
}

// ---------------------------------------------------------------------------
// K2: f1/f2 per (head, node)
// ---------------------------------------------------------------------------
__global__ void __launch_bounds__(256) k_logits1(
    const float* __restrict__ fts, const float* __restrict__ a1,
    const float* __restrict__ b1, const float* __restrict__ a2,
    const float* __restrict__ b2, float* __restrict__ f1,
    float* __restrict__ f2) {
  int tid = blockIdx.x * 256 + threadIdx.x;     // NH * N threads
  int head = tid >> 12, n = tid & 4095;
  const float* fr  = fts + n * 512 + head * 64;
  const float* a1h = a1 + head * 64;
  const float* a2h = a2 + head * 64;
  float s1 = 0.f, s2 = 0.f;
#pragma unroll 8
  for (int h = 0; h < H_DIM; ++h) { float v = fr[h]; s1 += v * a1h[h]; s2 += v * a2h[h]; }
  f1[tid] = s1 + b1[head];
  f2[tid] = s2 + b2[head];
}

// ---------------------------------------------------------------------------
// K3: fused 8-head attention. One block per 16-row tile; wave w = head w.
// 16x64 bias tiles, ping-pong LDS buffers, one barrier per 64 columns.
// No softmax rescale needed: logits are O(1); -1e9 mask underflows exp to 0.
// ---------------------------------------------------------------------------
__global__ void __launch_bounds__(256) k_flash1(
    const float* __restrict__ bias, const float* __restrict__ f1,
    const float* __restrict__ f2, const _Float16* __restrict__ ftsT,
    _Float16* __restrict__ vals16) {
  __shared__ float sbias[2][16 * 64];
  const int t = threadIdx.x;
  const int lane = t & 31, head = t >> 5;
  const int i0 = blockIdx.x * 16;
  const int mlo = lane & 15, hi = lane >> 4, kb = 16 * hi;
  const float f1v = f1[head * N_NODES + i0 + mlo];
  const float* f2h = f2 + head * N_NODES;
  const _Float16* bbase = ftsT + (head * 64 + mlo) * N_NODES + 16 * hi;
  float rowsum = 0.f;
  v8f acc0 = {}, acc1 = {}, acc2 = {}, acc3 = {};

  // stage first 16x64 bias tile
#pragma unroll
  for (int q = 0; q < 4; ++q) {
    int e = t + q * 256;
    sbias[0][e] = bias[(i0 + (e >> 6)) * N_NODES + (e & 63)];
  }
  __syncthreads();

  for (int jt = 0; jt < N_NODES / 64; ++jt) {
    const int j0 = jt * 64;
    const int p = jt & 1;
    if (jt + 1 < N_NODES / 64) {
#pragma unroll
      for (int q = 0; q < 4; ++q) {
        int e = t + q * 256;
        sbias[p ^ 1][e] = bias[(i0 + (e >> 6)) * N_NODES + j0 + 64 + (e & 63)];
      }
    }
    // two 32-column chunks from the staged tile
#pragma unroll
    for (int c = 0; c < 2; ++c) {
      const int jc = j0 + c * 32;
      const float* sb = &sbias[p][c * 32];
      v8f f2lo = *(const v8f*)(f2h + jc + kb);
      v8f f2hi = *(const v8f*)(f2h + jc + 8 + kb);
      v16h pr;
#pragma unroll
      for (int i = 0; i < 8; ++i) {
        float x = f1v + f2lo[i];
        x = x > 0.f ? x : 0.2f * x;
        float e = __expf(x + sb[mlo * 64 + kb + i]);
        rowsum += e; pr[i] = (_Float16)e;
        float x2 = f1v + f2hi[i];
        x2 = x2 > 0.f ? x2 : 0.2f * x2;
        float e2 = __expf(x2 + sb[mlo * 64 + 8 + kb + i]);
        rowsum += e2; pr[8 + i] = (_Float16)e2;
      }
      const _Float16* bp = bbase + jc;
      v16h b0 = *(const v16h*)(bp);
      v16h b1 = *(const v16h*)(bp + 16 * N_NODES);
      v16h b2 = *(const v16h*)(bp + 32 * N_NODES);
      v16h b3 = *(const v16h*)(bp + 48 * N_NODES);
      acc0 = wmma_f16(pr, b0, acc0);
      acc1 = wmma_f16(pr, b1, acc1);
      acc2 = wmma_f16(pr, b2, acc2);
      acc3 = wmma_f16(pr, b3, acc3);
    }
    __syncthreads();
  }
  rowsum += __shfl_xor(rowsum, 16, 32);   // combine the two K-halves of row mlo
  float inv = 1.f / rowsum;               // lane r (r<16) holds row r's 1/sum
#pragma unroll
  for (int v = 0; v < 8; ++v) {
    const int m = 8 * hi + v;
    const float im = __shfl(inv, m, 32);
    _Float16* o = vals16 + (i0 + m) * 512 + head * 64 + mlo;
    o[ 0] = (_Float16)(acc0[v] * im);
    o[16] = (_Float16)(acc1[v] * im);
    o[32] = (_Float16)(acc2[v] * im);
    o[48] = (_Float16)(acc3[v] * im);
  }
}

// ---------------------------------------------------------------------------
// K4: h1 = elu(vals @ W2 + bW); fully unrolled (K=64), loads up front.
// ---------------------------------------------------------------------------
__global__ void __launch_bounds__(256) k_gemm_w2(
    const _Float16* __restrict__ vals16, const _Float16* __restrict__ W2T,
    const float* __restrict__ bW, _Float16* __restrict__ h116) {
  const int lane = threadIdx.x & 31, head = threadIdx.x >> 5;
  const int i0 = blockIdx.x * 16;
  const int mlo = lane & 15, hi = lane >> 4;
  const _Float16* arow = vals16 + (i0 + mlo) * 512 + head * 64;
  const _Float16* bp   = W2T + (head * 64 + mlo) * H_DIM + 16 * hi;

  v16h a0  = load_a16(arow + 16 * hi, arow + 8 + 16 * hi);
  v16h a1f = load_a16(arow + 32 + 16 * hi, arow + 40 + 16 * hi);
  v16h b00 = *(const v16h*)(bp);
  v16h b01 = *(const v16h*)(bp + 16 * H_DIM);
  v16h b02 = *(const v16h*)(bp + 32 * H_DIM);
  v16h b03 = *(const v16h*)(bp + 48 * H_DIM);
  v16h b10 = *(const v16h*)(bp + 32);
  v16h b11 = *(const v16h*)(bp + 32 + 16 * H_DIM);
  v16h b12 = *(const v16h*)(bp + 32 + 32 * H_DIM);
  v16h b13 = *(const v16h*)(bp + 32 + 48 * H_DIM);

  v8f acc0 = {}, acc1 = {}, acc2 = {}, acc3 = {};
  acc0 = wmma_f16(a0, b00, acc0);
  acc1 = wmma_f16(a0, b01, acc1);
  acc2 = wmma_f16(a0, b02, acc2);
  acc3 = wmma_f16(a0, b03, acc3);
  acc0 = wmma_f16(a1f, b10, acc0);
  acc1 = wmma_f16(a1f, b11, acc1);
  acc2 = wmma_f16(a1f, b12, acc2);
  acc3 = wmma_f16(a1f, b13, acc3);

#pragma unroll
  for (int v = 0; v < 8; ++v) {
    const int m = 8 * hi + v;
    _Float16* o = h116 + (i0 + m) * 512 + head * 64 + mlo;
#pragma unroll
    for (int nt = 0; nt < 4; ++nt) {
      float c = (nt == 0 ? acc0[v] : nt == 1 ? acc1[v] : nt == 2 ? acc2[v] : acc3[v]);
      float x = c + bW[head * 64 + nt * 16 + mlo];
      float e = x > 0.f ? x : (__expf(x) - 1.f);
      o[nt * 16] = (_Float16)e;
    }
  }
}

// ---------------------------------------------------------------------------
// K5: fts_o = h1 @ Wo1 (4096x512x64, C padded to 64); pipelined k-loop.
// ---------------------------------------------------------------------------
__global__ void __launch_bounds__(256) k_gemm_ftso(
    const _Float16* __restrict__ h116, const _Float16* __restrict__ Wo1T,
    float* __restrict__ ftso, _Float16* __restrict__ ftsoT) {
  const int lane = threadIdx.x & 31, wave = threadIdx.x >> 5;
  const int i0 = blockIdx.x * 128 + wave * 16;
  const int mlo = lane & 15, hi = lane >> 4;
  v8f acc0 = {}, acc1 = {}, acc2 = {}, acc3 = {};
  const _Float16* arow = h116 + (i0 + mlo) * F_IN;
  const _Float16* brow = Wo1T + mlo * F_IN + 16 * hi;

  v16h a  = load_a16(arow + 16 * hi, arow + 8 + 16 * hi);
  v16h b0 = *(const v16h*)(brow);
  v16h b1 = *(const v16h*)(brow + 16 * F_IN);
  v16h b2 = *(const v16h*)(brow + 32 * F_IN);
  v16h b3 = *(const v16h*)(brow + 48 * F_IN);
#pragma unroll 3
  for (int kt = 0; kt < F_IN / 32 - 1; ++kt) {
    const int k1 = (kt + 1) * 32;
    v16h an  = load_a16(arow + k1 + 16 * hi, arow + k1 + 8 + 16 * hi);
    v16h bn0 = *(const v16h*)(brow + k1);
    v16h bn1 = *(const v16h*)(brow + k1 + 16 * F_IN);
    v16h bn2 = *(const v16h*)(brow + k1 + 32 * F_IN);
    v16h bn3 = *(const v16h*)(brow + k1 + 48 * F_IN);
    acc0 = wmma_f16(a, b0, acc0);
    acc1 = wmma_f16(a, b1, acc1);
    acc2 = wmma_f16(a, b2, acc2);
    acc3 = wmma_f16(a, b3, acc3);
    a = an; b0 = bn0; b1 = bn1; b2 = bn2; b3 = bn3;
  }
  acc0 = wmma_f16(a, b0, acc0);
  acc1 = wmma_f16(a, b1, acc1);
  acc2 = wmma_f16(a, b2, acc2);
  acc3 = wmma_f16(a, b3, acc3);

#pragma unroll
  for (int v = 0; v < 8; ++v) {
    const int m = 8 * hi + v, row = i0 + m;
    float c0 = acc0[v], c1 = acc1[v], c2 = acc2[v], c3 = acc3[v];
    ftso[row * C_PAD + mlo +  0] = c0;
    ftso[row * C_PAD + mlo + 16] = c1;
    ftso[row * C_PAD + mlo + 32] = c2;
    ftso[row * C_PAD + mlo + 48] = c3;
    ftsoT[(mlo +  0) * N_NODES + row] = (_Float16)c0;
    ftsoT[(mlo + 16) * N_NODES + row] = (_Float16)c1;
    ftsoT[(mlo + 32) * N_NODES + row] = (_Float16)c2;
    ftsoT[(mlo + 48) * N_NODES + row] = (_Float16)c3;
  }
}

// ---------------------------------------------------------------------------
// K6: output-layer logits
// ---------------------------------------------------------------------------
__global__ void __launch_bounds__(256) k_logits_o(
    const float* __restrict__ ftso, const float* __restrict__ ao1,
    const float* __restrict__ bo1, const float* __restrict__ ao2,
    const float* __restrict__ bo2, float* __restrict__ f1o,
    float* __restrict__ f2o) {
  int n = blockIdx.x * 256 + threadIdx.x;
  const float* fr = ftso + n * C_PAD;
  float s1 = 0.f, s2 = 0.f;
#pragma unroll 8
  for (int c = 0; c < C_CLS; ++c) { float v = fr[c]; s1 += v * ao1[c]; s2 += v * ao2[c]; }
  f1o[n] = s1 + bo1[0];
  f2o[n] = s2 + bo2[0];
}

// ---------------------------------------------------------------------------
// K7: output-layer attention; single head, bias L2-resident from k_flash1.
// 64 columns per iteration so chunk-1 loads overlap chunk-0 WMMAs.
// ---------------------------------------------------------------------------
__global__ void __launch_bounds__(256) k_flash2(
    const float* __restrict__ bias, const float* __restrict__ f1o,
    const float* __restrict__ f2o, const _Float16* __restrict__ ftsoT,
    float* __restrict__ valso) {
  const int lane = threadIdx.x & 31, wave = threadIdx.x >> 5;
  const int i0 = blockIdx.x * 128 + wave * 16;
  const int mlo = lane & 15, hi = lane >> 4, kb = 16 * hi;
  const float f1v = f1o[i0 + mlo];
  const float* brow = bias + (i0 + mlo) * N_NODES;
  const _Float16* bbase = ftsoT + mlo * N_NODES + 16 * hi;
  float rowsum = 0.f;
  v8f acc0 = {}, acc1 = {}, acc2 = {}, acc3 = {};
  for (int jt = 0; jt < N_NODES / 64; ++jt) {
#pragma unroll
    for (int c = 0; c < 2; ++c) {
      const int jc = jt * 64 + c * 32;
      v8f blo  = *(const v8f*)(brow + jc + kb);
      v8f bhi  = *(const v8f*)(brow + jc + 8 + kb);
      v8f f2lo = *(const v8f*)(f2o + jc + kb);
      v8f f2hi = *(const v8f*)(f2o + jc + 8 + kb);
      v16h pr;
#pragma unroll
      for (int i = 0; i < 8; ++i) {
        float x = f1v + f2lo[i];
        x = x > 0.f ? x : 0.2f * x;
        float e = __expf(x + blo[i]);
        rowsum += e; pr[i] = (_Float16)e;
        float x2 = f1v + f2hi[i];
        x2 = x2 > 0.f ? x2 : 0.2f * x2;
        float e2 = __expf(x2 + bhi[i]);
        rowsum += e2; pr[8 + i] = (_Float16)e2;
      }
      const _Float16* bp = bbase + jc;
      v16h b0 = *(const v16h*)(bp);
      v16h b1 = *(const v16h*)(bp + 16 * N_NODES);
      v16h b2 = *(const v16h*)(bp + 32 * N_NODES);
      v16h b3 = *(const v16h*)(bp + 48 * N_NODES);
      acc0 = wmma_f16(pr, b0, acc0);
      acc1 = wmma_f16(pr, b1, acc1);
      acc2 = wmma_f16(pr, b2, acc2);
      acc3 = wmma_f16(pr, b3, acc3);
    }
  }
  rowsum += __shfl_xor(rowsum, 16, 32);
  float inv = 1.f / rowsum;
#pragma unroll
  for (int v = 0; v < 8; ++v) {
    const int m = 8 * hi + v;
    const float im = __shfl(inv, m, 32);
    float* o = valso + (i0 + m) * C_PAD + mlo;
    o[ 0] = acc0[v] * im;
    o[16] = acc1[v] * im;
    o[32] = acc2[v] * im;
    o[48] = acc3[v] * im;
  }
}

// ---------------------------------------------------------------------------
// K8: out[n][c] = vals_o[n,:40] . Wo2[:,c] + bo[c]   (tiny, VALU)
// ---------------------------------------------------------------------------
__global__ void __launch_bounds__(256) k_final(
    const float* __restrict__ valso, const float* __restrict__ Wo2,
    const float* __restrict__ bo, float* __restrict__ out) {
  int tid = blockIdx.x * 256 + threadIdx.x;
  if (tid >= N_NODES * C_CLS) return;
  int n = tid / C_CLS, c = tid % C_CLS;
  const float* vr = valso + n * C_PAD;
  float s = 0.f;
#pragma unroll 8
  for (int k = 0; k < C_CLS; ++k) s += vr[k] * Wo2[k * C_CLS + c];
  out[tid] = s + bo[c];
}

// ---------------------------------------------------------------------------
extern "C" void kernel_launch(void* const* d_in, const int* in_sizes, int n_in,
                              void* d_out, int out_size, void* d_ws, size_t ws_size,
                              hipStream_t stream) {
  const float* seq  = (const float*)d_in[0];
  const float* bias = (const float*)d_in[1];
  const float* W1   = (const float*)d_in[2];
  const float* a1   = (const float*)d_in[3];
  const float* b1   = (const float*)d_in[4];
  const float* a2   = (const float*)d_in[5];
  const float* b2   = (const float*)d_in[6];
  const float* W2   = (const float*)d_in[7];
  const float* bW   = (const float*)d_in[8];
  const float* Wo1  = (const float*)d_in[9];
  const float* ao1  = (const float*)d_in[10];
  const float* bo1  = (const float*)d_in[11];
  const float* ao2  = (const float*)d_in[12];
  const float* bo2  = (const float*)d_in[13];
  const float* Wo2  = (const float*)d_in[14];
  const float* bo   = (const float*)d_in[15];
  float* out = (float*)d_out;

  char* ws = (char*)d_ws;
  size_t off = 0;
  auto alloc = [&](size_t bytes) { char* p = ws + off; off += (bytes + 255) & ~size_t(255); return p; };
  _Float16* seq16  = (_Float16*)alloc((size_t)N_NODES * F_IN * 2);
  _Float16* W1T    = (_Float16*)alloc((size_t)NH * H_DIM * F_IN * 2);
  _Float16* W2T    = (_Float16*)alloc((size_t)NH * H_DIM * H_DIM * 2);
  _Float16* Wo1T   = (_Float16*)alloc((size_t)C_PAD * F_IN * 2);
  float*    fts    = (float*)   alloc((size_t)N_NODES * 512 * 4);
  _Float16* ftsT   = (_Float16*)alloc((size_t)512 * N_NODES * 2);
  float*    f1     = (float*)   alloc((size_t)NH * N_NODES * 4);
  float*    f2     = (float*)   alloc((size_t)NH * N_NODES * 4);
  _Float16* vals16 = (_Float16*)alloc((size_t)N_NODES * 512 * 2);
  _Float16* h116   = (_Float16*)alloc((size_t)N_NODES * 512 * 2);
  float*    ftso   = (float*)   alloc((size_t)N_NODES * C_PAD * 4);
  _Float16* ftsoT  = (_Float16*)alloc((size_t)C_PAD * N_NODES * 2);
  float*    f1o    = (float*)   alloc((size_t)N_NODES * 4);
  float*    f2o    = (float*)   alloc((size_t)N_NODES * 4);
  float*    valso  = (float*)   alloc((size_t)N_NODES * C_PAD * 4);

  k_pack<<<(N_NODES * F_IN + 255) / 256, 256, 0, stream>>>(
      seq, W1, W2, Wo1, seq16, W1T, W2T, Wo1T);
  k_gemm_fts<<<N_NODES / 16, 256, 0, stream>>>(seq16, W1T, fts, ftsT);
  k_logits1<<<(NH * N_NODES) / 256, 256, 0, stream>>>(fts, a1, b1, a2, b2, f1, f2);
  k_flash1<<<N_NODES / 16, 256, 0, stream>>>(bias, f1, f2, ftsT, vals16);
  k_gemm_w2<<<N_NODES / 16, 256, 0, stream>>>(vals16, W2T, bW, h116);
  k_gemm_ftso<<<N_NODES / 128, 256, 0, stream>>>(h116, Wo1T, ftso, ftsoT);
  k_logits_o<<<N_NODES / 256, 256, 0, stream>>>(ftso, ao1, bo1, ao2, bo2, f1o, f2o);
  k_flash2<<<N_NODES / 128, 256, 0, stream>>>(bias, f1o, f2o, ftsoT, valso);
  k_final<<<(N_NODES * C_CLS + 255) / 256, 256, 0, stream>>>(valso, Wo2, bo, out);
}
